// GlmImageTextAttention_34694745817243
// MI455X (gfx1250) — matrix operand order repaired and checked by reading, not compiled, not run on hardware
//
#include <hip/hip_runtime.h>
#include <hip/hip_bf16.h>

#define B_ 2
#define S_ 2048
#define H_ 2048
#define NH_ 16
#define NKV_ 4
#define HD_ 128
#define NEGV (-1000000000.0f)

typedef __attribute__((ext_vector_type(16))) __bf16 v16bf;
typedef __attribute__((ext_vector_type(8)))  float  v8f;
typedef __attribute__((ext_vector_type(4)))  unsigned int u32x4;
typedef __attribute__((ext_vector_type(8)))  int i32x8;
typedef __attribute__((ext_vector_type(4)))  int i32x4;

union FragBF {
    v16bf v;
    unsigned int u[8];
};

__device__ __forceinline__ unsigned short f2bf(float f) {
    unsigned int u = __float_as_uint(f);
    return (unsigned short)((u + 0x7FFFu + ((u >> 16) & 1u)) >> 16);
}

// ---------------------------------------------------------------------------
// Async global->LDS copy of one 128-bit chunk (GV mode). ASYNCcnt-tracked.
// ---------------------------------------------------------------------------
__device__ __forceinline__ void async_copy_b128(unsigned lds_off,
                                                const void* gaddr) {
    unsigned long long ga = (unsigned long long)gaddr;
    asm volatile("global_load_async_to_lds_b128 %0, %1, off"
                 :: "v"(lds_off), "v"(ga) : "memory");
}
__device__ __forceinline__ void wait_async0() {
    asm volatile("s_wait_asynccnt 0" ::: "memory");
}

// ---------------------------------------------------------------------------
// TDM: load a 2D tile (tile_d0 x tile_d1 elements, 2-byte data) from a
// row-major tensor (row length stride_d0 elements) into LDS at lds_off.
// D# layout per CDNA5 ISA. TENSORcnt-tracked.
// Toolchain uses the 6-arg builtin: (u32x4, i32x8, i32x4, i32x4, i32x8, cpol).
// ---------------------------------------------------------------------------
#if __has_builtin(__builtin_amdgcn_tensor_load_to_lds)
#define HAVE_TDM 1
__device__ __forceinline__ void tdm_load_2d_bf16(unsigned lds_off,
                                                 const void* gptr,
                                                 unsigned tile_d0, unsigned tile_d1,
                                                 unsigned tensor_d0, unsigned tensor_d1,
                                                 unsigned stride_d0) {
    unsigned long long ga = (unsigned long long)gptr;
    u32x4 g0;
    g0.x = 1u;                                           // count=1, user D#
    g0.y = lds_off;                                      // lds_addr (bytes)
    g0.z = (unsigned)(ga & 0xFFFFFFFFull);               // global_addr[31:0]
    g0.w = (unsigned)((ga >> 32) & 0x1FFFFFFull)         // global_addr[56:32]
         | 0x80000000u;                                  // type = 2 ("image")
    i32x8 g1;
    g1[0] = (int)(1u << 16);                             // data_size=1 -> 2 bytes
    g1[1] = (int)((tensor_d0 & 0xFFFFu) << 16);          // tensor_dim0[15:0]
    g1[2] = (int)(((tensor_d0 >> 16) & 0xFFFFu)          // tensor_dim0[31:16]
                | ((tensor_d1 & 0xFFFFu) << 16));        // tensor_dim1[15:0]
    g1[3] = (int)(((tensor_d1 >> 16) & 0xFFFFu)          // tensor_dim1[31:16]
                | ((tile_d0 & 0xFFFFu) << 16));          // tile_dim0
    g1[4] = (int)(tile_d1 & 0xFFFFu);                    // tile_dim1 (tile_dim2=0)
    g1[5] = (int)stride_d0;                              // tensor_dim0_stride[31:0]
    g1[6] = 0;                                           // stride[47:32]=0, dim1_stride lo=0
    g1[7] = 0;
    i32x4 gz4 = {0, 0, 0, 0};
    i32x8 gz8 = {0, 0, 0, 0, 0, 0, 0, 0};
    __builtin_amdgcn_tensor_load_to_lds(g0, g1, gz4, gz4, gz8, 0);
}
#else
#define HAVE_TDM 0
#endif

// ---------------------------------------------------------------------------
// f32 -> bf16 conversion (grid-stride)
// ---------------------------------------------------------------------------
__global__ void f32_to_bf16_kernel(const float* __restrict__ in,
                                   unsigned short* __restrict__ out,
                                   size_t n) {
    size_t i = (size_t)blockIdx.x * blockDim.x + threadIdx.x;
    size_t stride = (size_t)gridDim.x * blockDim.x;
    for (; i < n; i += stride) out[i] = f2bf(in[i]);
}

// ---------------------------------------------------------------------------
// bf16 GEMM with f32 accumulate: C[M,N] = A[M,K] @ B[K,N] (+bias)
// 256 threads = 8 waves; 128x128 tile; each wave does 16(M) x 128(N).
// A tile staged by TDM; B tile staged transposed [n][k] so B fragments are
// contiguous u32/b128 reads.
// ---------------------------------------------------------------------------
__global__ __launch_bounds__(256)
void gemm_bf16_wmma_kernel(const unsigned short* __restrict__ A,
                           const unsigned short* __restrict__ Bm,
                           const float* __restrict__ bias,
                           float* __restrict__ C,
                           int M, int N, int K) {
    __shared__ __align__(16) unsigned short As[128][32];
    __shared__ __align__(16) unsigned short Bt[128][32];

    const int tid  = threadIdx.x;
    const int wave = tid >> 5;
    const int lane = tid & 31;
    const int hi   = lane >> 4;
    const int l16  = lane & 15;
    const int n0   = blockIdx.x * 128;
    const int m0   = blockIdx.y * 128;
    const unsigned asBase = (unsigned)(size_t)&As[0][0];

    v8f acc[8] = {};

    for (int k0 = 0; k0 < K; k0 += 32) {
#if HAVE_TDM
        // One TDM descriptor moves the whole 128x32 A tile into LDS.
        if (wave == 0) {
            tdm_load_2d_bf16(asBase,
                             A + (size_t)m0 * K + k0,
                             /*tile_d0=*/32, /*tile_d1=*/128,
                             /*tensor_d0=*/(unsigned)K, /*tensor_d1=*/(unsigned)M,
                             /*stride_d0=*/(unsigned)K);
            __builtin_amdgcn_s_wait_tensorcnt(0);
        }
#else
        // Async global->LDS: 512 b128 chunks, 2 per thread.
        for (int i = 0; i < 2; ++i) {
            int e   = tid + i * 256;      // 0..511
            int row = e >> 2;             // 4 x b128 per 32-elem row
            int c   = e & 3;
            async_copy_b128(asBase + (unsigned)(row * 32 + c * 16) * 2,
                            A + (size_t)(m0 + row) * K + k0 + c * 16);
        }
#endif
        // Stage B tile transposed (needs transpose -> manual path)
        for (int i = 0; i < 16; ++i) {
            int e  = tid + i * 256;    // 0..4095
            int kk = e >> 7;           // 0..31
            int nn = e & 127;
            Bt[nn][kk] = Bm[(size_t)(k0 + kk) * N + n0 + nn];
        }
        // Prefetch next B tile rows into L2 (global_prefetch_b8)
        if (k0 + 32 < K && tid < 32)
            __builtin_prefetch(Bm + (size_t)(k0 + 32 + tid) * N + n0, 0, 1);
#if !HAVE_TDM
        wait_async0();
#endif
        __syncthreads();

        // A fragment (16x32 bf16): lane holds row m = l16;
        // K halves: j<8 -> k=j+8*hi ; j>=8 -> k=j-8+16+8*hi  (pairs are u32)
        FragBF a;
        {
            const unsigned short* ar = &As[wave * 16 + l16][0];
            #pragma unroll
            for (int p = 0; p < 4; ++p)
                a.u[p] = *(const unsigned int*)&ar[2 * p + 8 * hi];
            #pragma unroll
            for (int p = 4; p < 8; ++p)
                a.u[p] = *(const unsigned int*)&ar[2 * p + 8 + 8 * hi];
        }

        #pragma unroll
        for (int t = 0; t < 8; ++t) {
            FragBF b;
            const unsigned short* br = &Bt[t * 16 + l16][0];
            #pragma unroll
            for (int p = 0; p < 8; ++p)
                b.u[p] = *(const unsigned int*)&br[2 * p + ((p >> 2) << 3) + 8 * hi];
            acc[t] = __builtin_amdgcn_wmma_f32_16x16x32_bf16(
                false, a.v, false, b.v, (short)0, acc[t], false, false);
        }
        __syncthreads();
    }

    for (int t = 0; t < 8; ++t) {
        int col = n0 + t * 16 + l16;
        float bv = bias ? bias[col] : 0.0f;
        #pragma unroll
        for (int r = 0; r < 8; ++r) {
            int row = m0 + wave * 16 + r + 8 * hi;
            C[(size_t)row * N + col] = acc[t][r] + bv;
        }
    }
}

// ---------------------------------------------------------------------------
// RoPE (interleaved) + split QKV into bf16 Q[B,NH,S,HD], K/V[B,NKV,S,HD]
// ---------------------------------------------------------------------------
__global__ void rope_split_kernel(const float* __restrict__ qkv,   // [B*S, 3072]
                                  const int* __restrict__ positions,
                                  unsigned short* __restrict__ Qb,
                                  unsigned short* __restrict__ Kb,
                                  unsigned short* __restrict__ Vb) {
    const int total = B_ * S_ * (NH_ + 2 * NKV_);
    int idx = blockIdx.x * blockDim.x + threadIdx.x;
    if (idx >= total) return;
    const int hh = idx % (NH_ + 2 * NKV_);
    const int s  = (idx / (NH_ + 2 * NKV_)) % S_;
    const int b  = idx / ((NH_ + 2 * NKV_) * S_);

    const float* src = qkv + (size_t)(b * S_ + s) * 3072 + hh * HD_;
    const float pos = (float)positions[b * S_ + s];
    const float LOG_THETA = 9.210340371976184f;   // ln(10000)

    if (hh < NH_ + NKV_) {
        unsigned short* dst;
        if (hh < NH_)
            dst = Qb + ((size_t)(b * NH_ + hh) * S_ + s) * HD_;
        else
            dst = Kb + ((size_t)(b * NKV_ + (hh - NH_)) * S_ + s) * HD_;
        #pragma unroll 4
        for (int i = 0; i < HD_ / 2; ++i) {
            float invf = __expf(-(2.0f * i / (float)HD_) * LOG_THETA);
            float ang  = pos * invf;
            float c = __cosf(ang), sn = __sinf(ang);
            float x1 = src[2 * i], x2 = src[2 * i + 1];
            dst[2 * i]     = f2bf(x1 * c - x2 * sn);
            dst[2 * i + 1] = f2bf(x2 * c + x1 * sn);
        }
    } else {
        unsigned short* dst = Vb + ((size_t)(b * NKV_ + (hh - NH_ - NKV_)) * S_ + s) * HD_;
        #pragma unroll 8
        for (int i = 0; i < HD_; ++i) dst[i] = f2bf(src[i]);
    }
}

// ---------------------------------------------------------------------------
// Flash-style causal GQA attention (4 waves per block; 64 Q rows per block).
// K tile staged via async global->LDS b128; V tile staged transposed.
// ---------------------------------------------------------------------------
__global__ __launch_bounds__(128)
void attn_fa_wmma_kernel(const unsigned short* __restrict__ Qb,
                         const unsigned short* __restrict__ Kb,
                         const unsigned short* __restrict__ Vb,
                         unsigned short* __restrict__ Ob) {   // [B, S, NH*HD] bf16
    __shared__ __align__(16) unsigned short Ks[32][128];
    __shared__ __align__(16) unsigned short Vt[128][32];
    __shared__ __align__(16) unsigned short Pl[4][16][32];

    const int tid = threadIdx.x;
    const int wv = tid >> 5, lane = tid & 31;
    const int hi = lane >> 4, l16 = lane & 15;
    const int qt = blockIdx.x;
    const int h  = blockIdx.y;
    const int b  = blockIdx.z;
    const int kvh = h >> 2;                 // NH/NKV = 4
    const int qbase = qt * 64;
    const int qrow0 = qbase + wv * 16;
    const unsigned ksBase = (unsigned)(size_t)&Ks[0][0];

    const unsigned short* Qp = Qb + (size_t)(b * NH_ + h)    * S_ * HD_;
    const unsigned short* Kp = Kb + (size_t)(b * NKV_ + kvh) * S_ * HD_;
    const unsigned short* Vp = Vb + (size_t)(b * NKV_ + kvh) * S_ * HD_;

    // Q fragments for 4 K-chunks of 32 (held in registers for the whole pass)
    FragBF aq[4];
    {
        const unsigned short* qr = Qp + (size_t)(qrow0 + l16) * HD_;
        #pragma unroll
        for (int c = 0; c < 4; ++c) {
            #pragma unroll
            for (int p = 0; p < 4; ++p)
                aq[c].u[p] = *(const unsigned int*)&qr[32 * c + 2 * p + 8 * hi];
            #pragma unroll
            for (int p = 4; p < 8; ++p)
                aq[c].u[p] = *(const unsigned int*)&qr[32 * c + 2 * p + 8 + 8 * hi];
        }
    }

    v8f o[8] = {};
    float mrow[8], lrow[8];
    #pragma unroll
    for (int r = 0; r < 8; ++r) { mrow[r] = -3.0e38f; lrow[r] = 0.0f; }
    const float scale = 0.08838834764831845f;   // 1/sqrt(128)

    for (int jb = 0; jb < qbase + 64; jb += 32) {
        // K tile: async global->LDS, 256 b128 chunks, 2 per thread
        for (int i = 0; i < 2; ++i) {
            int e  = tid + i * 128;       // 0..255
            int kv = e >> 3;              // 8 x b128 per 128-elem row
            int c  = e & 7;
            async_copy_b128(ksBase + (unsigned)(kv * 128 + c * 16) * 2,
                            Kp + (size_t)(jb + kv) * HD_ + c * 16);
        }
        // V tile transposed (manual)
        for (int i = 0; i < 32; ++i) {
            int e  = tid + i * 128;        // 0..4095
            int kv = e >> 7, hd = e & 127;
            Vt[hd][kv] = Vp[(size_t)(jb + kv) * HD_ + hd];
        }
        wait_async0();
        __syncthreads();

        if (jb <= qrow0 + 15) {   // wave has at least one unmasked column
            v8f s0 = {}, s1 = {};
            #pragma unroll
            for (int c = 0; c < 4; ++c) {
                FragBF kb0, kb1;
                const unsigned short* kr0 = &Ks[l16][0];
                const unsigned short* kr1 = &Ks[16 + l16][0];
                #pragma unroll
                for (int p = 0; p < 8; ++p) {
                    kb0.u[p] = *(const unsigned int*)&kr0[32 * c + 2 * p + ((p >> 2) << 3) + 8 * hi];
                    kb1.u[p] = *(const unsigned int*)&kr1[32 * c + 2 * p + ((p >> 2) << 3) + 8 * hi];
                }
                s0 = __builtin_amdgcn_wmma_f32_16x16x32_bf16(
                    false, aq[c].v, false, kb0.v, (short)0, s0, false, false);
                s1 = __builtin_amdgcn_wmma_f32_16x16x32_bf16(
                    false, aq[c].v, false, kb1.v, (short)0, s1, false, false);
            }

            #pragma unroll
            for (int r = 0; r < 8; ++r) {
                int qi = qrow0 + r + 8 * hi;
                float v0 = s0[r] * scale;
                float v1 = s1[r] * scale;
                if (jb + l16 > qi)      v0 = NEGV;
                if (jb + 16 + l16 > qi) v1 = NEGV;
                float rmax = fmaxf(v0, v1);
                #pragma unroll
                for (int off = 8; off >= 1; off >>= 1)
                    rmax = fmaxf(rmax, __shfl_xor(rmax, off, 32));
                float mnew = fmaxf(mrow[r], rmax);
                float corr = __expf(mrow[r] - mnew);
                float p0 = __expf(v0 - mnew);
                float p1 = __expf(v1 - mnew);
                float rs = p0 + p1;
                #pragma unroll
                for (int off = 8; off >= 1; off >>= 1)
                    rs += __shfl_xor(rs, off, 32);
                lrow[r] = lrow[r] * corr + rs;
                mrow[r] = mnew;
                #pragma unroll
                for (int t = 0; t < 8; ++t) o[t][r] *= corr;
                int m = r + 8 * hi;
                Pl[wv][m][l16]      = f2bf(p0);
                Pl[wv][m][16 + l16] = f2bf(p1);
            }
            // C-layout -> A-layout re-stripe of P goes through LDS (same wave)
            asm volatile("s_wait_dscnt 0" ::: "memory");
            FragBF ap;
            {
                const unsigned short* pr = &Pl[wv][l16][0];
                #pragma unroll
                for (int p = 0; p < 4; ++p)
                    ap.u[p] = *(const unsigned int*)&pr[2 * p + 8 * hi];
                #pragma unroll
                for (int p = 4; p < 8; ++p)
                    ap.u[p] = *(const unsigned int*)&pr[2 * p + 8 + 8 * hi];
            }
            #pragma unroll
            for (int t = 0; t < 8; ++t) {
                FragBF vb;
                const unsigned short* vr = &Vt[t * 16 + l16][0];
                #pragma unroll
                for (int p = 0; p < 8; ++p)
                    vb.u[p] = *(const unsigned int*)&vr[2 * p + ((p >> 2) << 3) + 8 * hi];
                o[t] = __builtin_amdgcn_wmma_f32_16x16x32_bf16(
                    false, ap.v, false, vb.v, (short)0, o[t], false, false);
            }
        }
        __syncthreads();
    }

    #pragma unroll
    for (int r = 0; r < 8; ++r) {
        float inv = 1.0f / lrow[r];
        int row = qrow0 + r + 8 * hi;
        unsigned short* dst = Ob + ((size_t)(b * S_ + row)) * (NH_ * HD_) + h * HD_;
        #pragma unroll
        for (int t = 0; t < 8; ++t)
            dst[t * 16 + l16] = f2bf(o[t][r] * inv);
    }
}

// ---------------------------------------------------------------------------
// Host-side launch
// ---------------------------------------------------------------------------
extern "C" void kernel_launch(void* const* d_in, const int* in_sizes, int n_in,
                              void* d_out, int out_size, void* d_ws, size_t ws_size,
                              hipStream_t stream) {
    (void)in_sizes; (void)n_in; (void)out_size; (void)ws_size;
    const float* hidden = (const float*)d_in[0];   // [B,S,H]
    const float* W_qkv  = (const float*)d_in[1];   // [H, 3072]
    const float* b_qkv  = (const float*)d_in[2];   // [3072]
    const float* W_o    = (const float*)d_in[3];   // [2048, 2048]
    const int*   pos    = (const int*)d_in[4];     // [B,S]

    const int M   = B_ * S_;                 // 4096
    const int Nq  = (NH_ + 2 * NKV_) * HD_;  // 3072
    const int K   = H_;                      // 2048

    char* ws = (char*)d_ws;
    float*          qkv_f32 = (float*)(ws + 0);                  // 50,331,648 B
    unsigned short* attn_bf = (unsigned short*)(ws + 0);         // reuses qkv region
    unsigned short* Abf     = (unsigned short*)(ws + 50331648);
    unsigned short* Wqkv_bf = (unsigned short*)(ws + 67108864);
    unsigned short* Wo_bf   = (unsigned short*)(ws + 79691776);
    unsigned short* Qb      = (unsigned short*)(ws + 88080384);
    unsigned short* Kb      = (unsigned short*)(ws + 104857600);
    unsigned short* Vb      = (unsigned short*)(ws + 109051904);

    f32_to_bf16_kernel<<<4096, 256, 0, stream>>>(hidden, Abf, (size_t)M * K);
    f32_to_bf16_kernel<<<4096, 256, 0, stream>>>(W_qkv, Wqkv_bf, (size_t)K * Nq);
    f32_to_bf16_kernel<<<4096, 256, 0, stream>>>(W_o, Wo_bf, (size_t)K * H_);

    {
        dim3 grid(Nq / 128, M / 128);
        gemm_bf16_wmma_kernel<<<grid, 256, 0, stream>>>(Abf, Wqkv_bf, b_qkv,
                                                        qkv_f32, M, Nq, K);
    }
    {
        int total = B_ * S_ * (NH_ + 2 * NKV_);
        rope_split_kernel<<<(total + 255) / 256, 256, 0, stream>>>(
            qkv_f32, pos, Qb, Kb, Vb);
    }
    {
        dim3 grid(S_ / 64, NH_, B_);
        attn_fa_wmma_kernel<<<grid, 128, 0, stream>>>(Qb, Kb, Vb, attn_bf);
    }
    {
        dim3 grid(H_ / 128, M / 128);
        gemm_bf16_wmma_kernel<<<grid, 256, 0, stream>>>(attn_bf, Wo_bf, nullptr,
                                                        (float*)d_out, M, H_, K);
    }
}
